// GIN_node_Virtual_17781164606040
// MI455X (gfx1250) — matrix operand order repaired and checked by reading, not compiled, not run on hardware
//
#include <hip/hip_runtime.h>

// ---------------------------------------------------------------------------
// GIN + virtual node forward for MI455X (gfx1250, wave32, WMMA).
// GEMMs run as bf16 WMMA (v_wmma_f32_16x16x32_bf16) with f32 accumulation.
// A-tiles staged via GLOBAL_LOAD_ASYNC_TO_LDS_B128 (ASYNCcnt path).
// Everything feeding BatchNorm batch statistics stays in f32.
// ---------------------------------------------------------------------------

typedef __attribute__((ext_vector_type(16))) __bf16 v16bf;
typedef __attribute__((ext_vector_type(8)))  __bf16 v8bf;
typedef __attribute__((ext_vector_type(8)))  float  v8f;
typedef __attribute__((ext_vector_type(4)))  int    v4i;

#define EMB   512
#define EMB2  1024
#define TM    128
#define TN    128
#define TK    32

#if defined(__AMDGCN__) && defined(__has_builtin)
#if __has_builtin(__builtin_amdgcn_global_load_async_to_lds_b128) && \
    __has_builtin(__builtin_amdgcn_s_wait_asynccnt)
#define USE_ASYNC_LDS 1
#endif
#endif
#ifndef USE_ASYNC_LDS
#define USE_ASYNC_LDS 0
#endif

// ---------------------------------------------------------------------------
// WMMA GEMM: C[M,N] = A[M,K](bf16,row) * B[K,N](bf16,row) + bias[N]
// Requires M%128==0, N%128==0, K%32==0 (true for all GEMMs in this net).
// Block = 256 threads = 8 waves; block tile 128x128; wave tile 64x32 (4x2 WMMA)
// ---------------------------------------------------------------------------
__global__ __launch_bounds__(256, 2)
void gemm_bf16_wmma(const __bf16* __restrict__ A, const __bf16* __restrict__ B,
                    const float* __restrict__ bias, float* __restrict__ C,
                    int M, int N, int K)
{
    __shared__ __attribute__((aligned(16))) __bf16 As[TM][TK];      // 8 KB row-major
    __shared__ __attribute__((aligned(16))) __bf16 Bs[TN][TK + 16]; // 12 KB transposed, 96B pitch
    const int tid  = threadIdx.x;
    const int wave = tid >> 5;
    const int lane = tid & 31;
    const int bm = blockIdx.y * TM;
    const int bn = blockIdx.x * TN;
    const int wm = (wave >> 2) * 64;       // wave row offset inside block tile
    const int wn = (wave & 3) * 32;        // wave col offset inside block tile
    const int lrow  = lane & 15;
    const int khalf = lane >> 4;

    v8f acc[4][2];
#pragma unroll
    for (int i = 0; i < 4; ++i)
#pragma unroll
        for (int j = 0; j < 2; ++j)
#pragma unroll
            for (int q = 0; q < 8; ++q) acc[i][j][q] = 0.f;

    for (int k0 = 0; k0 < K; k0 += TK) {
        // ---- stage A tile: 128x32 bf16, 32 bytes per thread ----
        {
            const int r  = tid >> 1;
            const int ks = (tid & 1) * 16;
            const __bf16* src = A + (size_t)(bm + r) * K + k0 + ks;
#if USE_ASYNC_LDS
            // async DMA into LDS (imm offset applies to both addresses)
            __builtin_amdgcn_global_load_async_to_lds_b128(
                (__attribute__((address_space(1))) v4i*)(__UINTPTR_TYPE__)(const void*)src,
                (__attribute__((address_space(3))) v4i*)&As[r][ks], 0, 0);
            __builtin_amdgcn_global_load_async_to_lds_b128(
                (__attribute__((address_space(1))) v4i*)(__UINTPTR_TYPE__)(const void*)src,
                (__attribute__((address_space(3))) v4i*)&As[r][ks], 16, 0);
#else
            *(v8bf*)&As[r][ks]     = *(const v8bf*)src;
            *(v8bf*)&As[r][ks + 8] = *(const v8bf*)(src + 8);
#endif
        }
        // ---- stage B tile transposed: 32x128 -> Bs[col][k] ----
        {
            const int kk  = tid >> 3;
            const int seg = (tid & 7) * 16;
            const __bf16* src = B + (size_t)(k0 + kk) * N + bn + seg;
            v8bf b0 = *(const v8bf*)src;
            v8bf b1 = *(const v8bf*)(src + 8);
#pragma unroll
            for (int j = 0; j < 8; ++j) Bs[seg + j][kk]     = b0[j];
#pragma unroll
            for (int j = 0; j < 8; ++j) Bs[seg + 8 + j][kk] = b1[j];
        }
#if USE_ASYNC_LDS
        __builtin_amdgcn_s_wait_asynccnt(0);
#endif
        __syncthreads();

        union Frag { v16bf v; v8bf h[2]; };
        Frag afrag[4], bfrag[2];
        // A frag: lanes 0-15 row M=lrow, K 0-15; lanes 16-31 K 16-31
#pragma unroll
        for (int i = 0; i < 4; ++i) {
            const __bf16* ap = &As[wm + i * 16 + lrow][khalf * 16];
            afrag[i].h[0] = *(const v8bf*)ap;
            afrag[i].h[1] = *(const v8bf*)(ap + 8);
        }
        // B frag: lanes 0-15 K=0..15, lanes 16-31 K=16..31, col = lane&15
#pragma unroll
        for (int j = 0; j < 2; ++j) {
            const __bf16* bp = &Bs[wn + j * 16 + lrow][khalf * 16];
            bfrag[j].h[0] = *(const v8bf*)bp;
            bfrag[j].h[1] = *(const v8bf*)(bp + 8);
        }
#if defined(__AMDGCN__)
#pragma unroll
        for (int i = 0; i < 4; ++i)
#pragma unroll
            for (int j = 0; j < 2; ++j)
                acc[i][j] = __builtin_amdgcn_wmma_f32_16x16x32_bf16(
                    false, afrag[i].v, false, bfrag[j].v,
                    (short)0, acc[i][j], false, false);
#endif
        __syncthreads();
    }

    // store D (lane 0-15: M=v, N=lane; lane 16-31: M=v+8, N=lane-16) + bias
#pragma unroll
    for (int i = 0; i < 4; ++i) {
        const int row0 = bm + wm + i * 16 + khalf * 8;
#pragma unroll
        for (int j = 0; j < 2; ++j) {
            const int col = bn + wn + j * 16 + lrow;
            const float bv = bias ? bias[col] : 0.f;
#pragma unroll
            for (int v = 0; v < 8; ++v)
                C[(size_t)(row0 + v) * N + col] = acc[i][j][v] + bv;
        }
    }
}

// ---------------------------------------------------------------------------
// elementwise / encoder / scatter kernels
// ---------------------------------------------------------------------------
__global__ void zero_f32(float* p, size_t n) {
    size_t i = (size_t)blockIdx.x * blockDim.x + threadIdx.x;
    if (i < n) p[i] = 0.f;
}

__global__ void f32_to_bf16_k(const float* __restrict__ s, __bf16* __restrict__ d, size_t n) {
    size_t i = (size_t)blockIdx.x * blockDim.x + threadIdx.x;
    if (i < n) d[i] = (__bf16)s[i];
}

// atom encoder: one-hot @ W == sum of 9 gathered embedding rows + bias
__global__ void atom_encode(const int* __restrict__ x, const float* __restrict__ W,
                            const float* __restrict__ b, float* __restrict__ h, int N) {
    size_t gid = (size_t)blockIdx.x * blockDim.x + threadIdx.x;
    if (gid >= (size_t)N * EMB) return;
    const int n = (int)(gid >> 9), c = (int)(gid & (EMB - 1));
    const int off[9] = {0, 119, 123, 135, 147, 157, 163, 169, 171};
    float s = b[c];
#pragma unroll
    for (int i = 0; i < 9; ++i)
        s += W[(size_t)(off[i] + x[n * 9 + i]) * EMB + c];
    h[gid] = s;
}

// bond encoder -> bf16 edge embeddings
__global__ void bond_encode(const int* __restrict__ ea, const float* __restrict__ W,
                            const float* __restrict__ b, __bf16* __restrict__ ee, int E) {
    size_t gid = (size_t)blockIdx.x * blockDim.x + threadIdx.x;
    if (gid >= (size_t)E * EMB) return;
    const int e = (int)(gid >> 9), c = (int)(gid & (EMB - 1));
    const int off[3] = {0, 5, 11};
    float s = b[c];
#pragma unroll
    for (int i = 0; i < 3; ++i)
        s += W[(size_t)(off[i] + ea[e * 3 + i]) * EMB + c];
    ee[gid] = (__bf16)s;
}

// t = h + vn[batch]
__global__ void add_vn_k(const float* __restrict__ h, const float* __restrict__ vn,
                         const int* __restrict__ batch, float* __restrict__ t, int rows) {
    size_t gid = (size_t)blockIdx.x * blockDim.x + threadIdx.x;
    if (gid >= (size_t)rows * EMB) return;
    const int n = (int)(gid >> 9), c = (int)(gid & (EMB - 1));
    t[gid] = h[gid] + vn[(size_t)batch[n] * EMB + c];
}

// aggr[dst] += relu(t[src] + eemb[e])
__global__ void scatter_msg(const float* __restrict__ t, const __bf16* __restrict__ ee,
                            const int* __restrict__ src, const int* __restrict__ dst,
                            float* __restrict__ aggr, int E) {
    size_t gid = (size_t)blockIdx.x * blockDim.x + threadIdx.x;
    if (gid >= (size_t)E * (EMB / 4)) return;
    const int e = (int)(gid >> 7);
    const int c = (int)(gid & 127) * 4;
    const int s = src[e], d = dst[e];
    const float*  tp = t    + (size_t)s * EMB + c;
    const __bf16* ep = ee   + (size_t)e * EMB + c;
    float*        ap = aggr + (size_t)d * EMB + c;
#pragma unroll
    for (int k = 0; k < 4; ++k) {
        float m = tp[k] + (float)ep[k];
        m = m > 0.f ? m : 0.f;
        atomicAdd(ap + k, m);
    }
}

// out_bf16 = (1+eps)*t + aggr
__global__ void gin_combine(const float* __restrict__ t, const float* __restrict__ aggr,
                            const float* __restrict__ epsp, __bf16* __restrict__ out, size_t n) {
    size_t i = (size_t)blockIdx.x * blockDim.x + threadIdx.x;
    if (i >= n) return;
    const float e = 1.f + *epsp;
    out[i] = (__bf16)(e * t[i] + aggr[i]);
}

// per-column sum & sumsq partials (batch-stats BatchNorm)
__global__ void colstats(const float* __restrict__ x, int M, int Nc,
                         float* __restrict__ sum, float* __restrict__ sumsq) {
    const int c = blockIdx.x * blockDim.x + threadIdx.x;
    if (c >= Nc) return;
    const int r0 = blockIdx.y * 4096;
    const int r1 = (r0 + 4096 < M) ? r0 + 4096 : M;
    float s = 0.f, s2 = 0.f;
    for (int r = r0; r < r1; ++r) {
        const float v = x[(size_t)r * Nc + c];
        s += v; s2 += v * v;
    }
    atomicAdd(&sum[c], s);
    atomicAdd(&sumsq[c], s2);
}

// y = g*(x-mean)*rsqrt(var+eps)+b, optional relu / row-mask-zero / f32 & bf16 out
__global__ void bn_apply(const float* __restrict__ x, const float* __restrict__ sum,
                         const float* __restrict__ sumsq, const float* __restrict__ g,
                         const float* __restrict__ b, int M, int Nc, int do_relu,
                         float* __restrict__ f32out, __bf16* __restrict__ bf16out,
                         const int* __restrict__ mask) {
    size_t gid = (size_t)blockIdx.x * blockDim.x + threadIdx.x;
    if (gid >= (size_t)M * Nc) return;
    const size_t r = gid / (size_t)Nc;
    const int    c = (int)(gid - r * Nc);
    const float invM = 1.f / (float)M;
    const float mean = sum[c] * invM;
    const float var  = sumsq[c] * invM - mean * mean;
    float v = g[c] * (x[gid] - mean) * rsqrtf(var + 1e-5f) + b[c];
    if (do_relu) v = fmaxf(v, 0.f);
    if (mask && mask[r]) v = 0.f;
    if (f32out)  f32out[gid]  = v;
    if (bf16out) bf16out[gid] = (__bf16)v;
}

// out[batch[r], coloff + c] += x[r, c]   (ldo = row pitch of out)
__global__ void seg_sum(const float* __restrict__ x, const int* __restrict__ batch,
                        float* __restrict__ out, int rows, int ldo, int coloff) {
    size_t gid = (size_t)blockIdx.x * blockDim.x + threadIdx.x;
    if (gid >= (size_t)rows * (EMB / 4)) return;
    const int r = (int)(gid >> 7);
    const int c = (int)(gid & 127) * 4;
    const int gg = batch[r];
    const float* xp = x + (size_t)r * EMB + c;
    float* op = out + (size_t)gg * ldo + coloff + c;
#pragma unroll
    for (int k = 0; k < 4; ++k) atomicAdd(op + k, xp[k]);
}

__global__ void agg_add_vn(float* __restrict__ agg, const float* __restrict__ vn, int G) {
    size_t gid = (size_t)blockIdx.x * blockDim.x + threadIdx.x;
    if (gid >= (size_t)G * EMB) return;
    const int gg = (int)(gid >> 9), c = (int)(gid & (EMB - 1));
    const float v = vn[gid];
    agg[(size_t)gg * EMB2 + c]       += v;
    agg[(size_t)gg * EMB2 + EMB + c] += v;
}

__global__ void count_batch(const int* __restrict__ batch, float* __restrict__ cnt, int n) {
    int i = blockIdx.x * blockDim.x + threadIdx.x;
    if (i < n) atomicAdd(&cnt[batch[i]], 1.f);
}

__global__ void pool_build(const float* __restrict__ s1, const float* __restrict__ s2,
                           const float* __restrict__ cnt, const int* __restrict__ nf,
                           float* __restrict__ agg, int G) {
    size_t gid = (size_t)blockIdx.x * blockDim.x + threadIdx.x;
    if (gid >= (size_t)G * EMB) return;
    const int gg = (int)(gid >> 9), c = (int)(gid & (EMB - 1));
    agg[(size_t)gg * EMB2 + c]       = s1[gid] / cnt[gg];
    agg[(size_t)gg * EMB2 + EMB + c] = s2[gid] / (float)nf[gg];
}

__global__ void broadcast_row(const float* __restrict__ row, float* __restrict__ dst, size_t rows) {
    size_t gid = (size_t)blockIdx.x * blockDim.x + threadIdx.x;
    if (gid < rows * EMB) dst[gid] = row[gid & (EMB - 1)];
}

// face encoder collapses (zero input + train-mode BN) to: relu(beta) @ W2 + b2
__global__ void face_init_row(const float* __restrict__ beta, const float* __restrict__ W2,
                              const float* __restrict__ b2, float* __restrict__ row) {
    const int c = blockIdx.x * blockDim.x + threadIdx.x;
    if (c >= EMB) return;
    float s = b2[c];
    for (int k = 0; k < EMB; ++k) {
        float r = beta[k];
        r = r > 0.f ? r : 0.f;
        s += r * W2[(size_t)k * EMB + c];
    }
    row[c] = s;
}

__global__ void scan_faces(const int* __restrict__ nf, int G, int* __restrict__ starts) {
    if (blockIdx.x == 0 && threadIdx.x == 0) {
        int a = 0;
        for (int g = 0; g < G; ++g) { starts[g] = a; a += nf[g]; }
        starts[G] = a;
    }
}

__global__ void fill_face_batch(const int* __restrict__ starts, const int* __restrict__ nf,
                                int* __restrict__ fb) {
    const int g = blockIdx.x;
    for (int i = threadIdx.x; i < nf[g]; i += blockDim.x) fb[starts[g] + i] = g;
}

// final head: out[g] = x[g,:] . w + b   (N-dim of 1, no WMMA needed)
__global__ void final_matvec(const float* __restrict__ x, const float* __restrict__ w,
                             const float* __restrict__ b, float* __restrict__ out, int G) {
    const int g = blockIdx.x * blockDim.x + threadIdx.x;
    if (g >= G) return;
    float s = b[0];
    for (int c = 0; c < EMB; ++c) s += x[(size_t)g * EMB + c] * w[c];
    out[g] = s;
}

// ---------------------------------------------------------------------------
// host orchestration
// ---------------------------------------------------------------------------
static inline unsigned nblk(size_t n, int t) { return (unsigned)((n + t - 1) / t); }

extern "C" void kernel_launch(void* const* d_in, const int* in_sizes, int n_in,
                              void* d_out, int out_size, void* d_ws, size_t ws_size,
                              hipStream_t stream) {
    (void)n_in; (void)ws_size; (void)out_size;
    const int N  = in_sizes[0] / 9;    // nodes
    const int E  = in_sizes[1] / 3;    // edges
    const int Fn = in_sizes[4];        // faces
    const int G  = in_sizes[6];        // graphs

    const int* x          = (const int*)d_in[0];
    const int* edge_attr  = (const int*)d_in[1];
    const int* ei_src     = (const int*)d_in[2];
    const int* ei_dst     = ei_src + E;
    const int* node_batch = (const int*)d_in[3];
    const int* face_mask  = (const int*)d_in[4];
    const int* fi_src     = (const int*)d_in[5];
    const int* fi_dst     = fi_src + E;
    const int* num_faces  = (const int*)d_in[6];

    // param pytree leaves (sorted dict keys, jax flatten order)
    auto P = [&](int i) { return (const float*)d_in[i]; };
    // 7 atom_W, 8 atom_b, 9..14 face_enc{W1,W2,b,b1,b2,g}
    // 15+20l layers{bn_b,bn_g,bnf_b,bnf_g,bond_W,bond_b,
    //               conv{W1,W2,b1,b2,bb1,g1}, conv_f{...}, eps, eps_f}
    // 115..120 pred{W1,W2,b,b1,b2,g}, 121 vn_emb, 122+8l vn_mlps{W1,W2,b1,b1n,b2,b2n,g1,g2}

    // workspace carve
    char* wsp = (char*)d_ws;
    auto carve = [&](size_t bytes) -> void* {
        void* p = wsp; wsp += (bytes + 255) & ~(size_t)255; return p;
    };
    float*  h      = (float*)carve((size_t)N * EMB * 4);
    float*  face   = (float*)carve((size_t)Fn * EMB * 4);
    float*  vnb    = (float*)carve((size_t)G * EMB * 4);
    float*  tbuf   = (float*)carve((size_t)N * EMB * 4);
    float*  aggr   = (float*)carve((size_t)N * EMB * 4);
    float*  mid    = (float*)carve((size_t)N * EMB2 * 4);
    __bf16* abf    = (__bf16*)carve((size_t)N * EMB2 * 2);
    __bf16* wbf    = (__bf16*)carve((size_t)EMB2 * EMB2 * 2);
    __bf16* eembbf = (__bf16*)carve((size_t)E * EMB * 2);
    float*  stats  = (float*)carve(2 * EMB2 * 4);
    float*  aggG   = (float*)carve((size_t)G * EMB2 * 4);
    float*  vtmp   = (float*)carve((size_t)G * EMB * 4);
    float*  s1     = (float*)carve((size_t)G * EMB * 4);
    float*  s2     = (float*)carve((size_t)G * EMB * 4);
    float*  cnt    = (float*)carve((size_t)G * 4);
    float*  frow   = (float*)carve(EMB * 4);
    int*    starts = (int*)carve((size_t)(G + 1) * 4);
    int*    fb     = (int*)carve((size_t)Fn * 4);

    auto gemm = [&](const __bf16* A, const float* Wf, const float* bias,
                    float* Cout, int M, int Nn, int Kk) {
        const size_t wn = (size_t)Kk * Nn;
        f32_to_bf16_k<<<nblk(wn, 256), 256, 0, stream>>>(Wf, wbf, wn);
        dim3 grid(Nn / TN, M / TM);
        gemm_bf16_wmma<<<grid, 256, 0, stream>>>(A, wbf, bias, Cout, M, Nn, Kk);
    };
    auto bn = [&](const float* xin, const float* gamma, const float* beta, int M, int Nc,
                  bool relu, float* f32out, __bf16* bf16out, const int* mask) {
        zero_f32<<<nblk(2 * Nc, 256), 256, 0, stream>>>(stats, 2 * Nc);
        dim3 gs(nblk(Nc, 256), nblk(M, 4096));
        colstats<<<gs, 256, 0, stream>>>(xin, M, Nc, stats, stats + Nc);
        const size_t tot = (size_t)M * Nc;
        bn_apply<<<nblk(tot, 256), 256, 0, stream>>>(xin, stats, stats + Nc, gamma, beta,
                                                     M, Nc, relu ? 1 : 0, f32out, bf16out, mask);
    };

    // ---- encoders & state init ----
    atom_encode<<<nblk((size_t)N * EMB, 256), 256, 0, stream>>>(x, P(7), P(8), h, N);
    broadcast_row<<<nblk((size_t)G * EMB, 256), 256, 0, stream>>>(P(121), vnb, G);
    scan_faces<<<1, 1, 0, stream>>>(num_faces, G, starts);
    fill_face_batch<<<G, 32, 0, stream>>>(starts, num_faces, fb);
    face_init_row<<<nblk(EMB, 256), 256, 0, stream>>>(P(11), P(10), P(13), frow);
    broadcast_row<<<nblk((size_t)Fn * EMB, 256), 256, 0, stream>>>(frow, face, Fn);

    // ---- layers ----
    for (int l = 0; l < 5; ++l) {
        const int L = 15 + 20 * l;
        const bool relu = (l < 4);
        bond_encode<<<nblk((size_t)E * EMB, 256), 256, 0, stream>>>(
            edge_attr, P(L + 4), P(L + 5), eembbf, E);

        // node GIN conv
        add_vn_k<<<nblk((size_t)N * EMB, 256), 256, 0, stream>>>(h, vnb, node_batch, tbuf, N);
        zero_f32<<<nblk((size_t)N * EMB, 256), 256, 0, stream>>>(aggr, (size_t)N * EMB);
        scatter_msg<<<nblk((size_t)E * 128, 256), 256, 0, stream>>>(
            tbuf, eembbf, ei_src, ei_dst, aggr, E);
        gin_combine<<<nblk((size_t)N * EMB, 256), 256, 0, stream>>>(
            tbuf, aggr, P(L + 18), abf, (size_t)N * EMB);
        gemm(abf, P(L + 6), P(L + 8), mid, N, EMB2, EMB);             // W1 + b1
        bn(mid, P(L + 11), P(L + 10), N, EMB2, true, nullptr, abf, nullptr); // g1, bb1
        gemm(abf, P(L + 7), P(L + 9), tbuf, N, EMB, EMB2);            // W2 + b2
        bn(tbuf, P(L + 1), P(L + 0), N, EMB, relu, h, nullptr, nullptr);     // bn_g, bn_b

        // face GIN conv
        add_vn_k<<<nblk((size_t)Fn * EMB, 256), 256, 0, stream>>>(face, vnb, fb, tbuf, Fn);
        zero_f32<<<nblk((size_t)Fn * EMB, 256), 256, 0, stream>>>(aggr, (size_t)Fn * EMB);
        scatter_msg<<<nblk((size_t)E * 128, 256), 256, 0, stream>>>(
            tbuf, eembbf, fi_src, fi_dst, aggr, E);
        gin_combine<<<nblk((size_t)Fn * EMB, 256), 256, 0, stream>>>(
            tbuf, aggr, P(L + 19), abf, (size_t)Fn * EMB);
        gemm(abf, P(L + 12), P(L + 14), mid, Fn, EMB2, EMB);
        bn(mid, P(L + 17), P(L + 16), Fn, EMB2, true, nullptr, abf, nullptr);
        gemm(abf, P(L + 13), P(L + 15), tbuf, Fn, EMB, EMB2);
        bn(tbuf, P(L + 3), P(L + 2), Fn, EMB, relu, face, nullptr, face_mask); // + where(mask,0)

        // virtual node update
        if (l < 4) {
            const int M8 = 122 + 8 * l;
            zero_f32<<<nblk((size_t)G * EMB2, 256), 256, 0, stream>>>(aggG, (size_t)G * EMB2);
            seg_sum<<<nblk((size_t)N * 128, 256), 256, 0, stream>>>(h, node_batch, aggG, N, EMB2, 0);
            seg_sum<<<nblk((size_t)Fn * 128, 256), 256, 0, stream>>>(face, fb, aggG, Fn, EMB2, EMB);
            agg_add_vn<<<nblk((size_t)G * EMB, 256), 256, 0, stream>>>(aggG, vnb, G);
            f32_to_bf16_k<<<nblk((size_t)G * EMB2, 256), 256, 0, stream>>>(aggG, abf, (size_t)G * EMB2);
            gemm(abf, P(M8 + 0), P(M8 + 2), vtmp, G, EMB, EMB2);      // W1 + b1
            bn(vtmp, P(M8 + 6), P(M8 + 3), G, EMB, true, nullptr, abf, nullptr); // g1, b1n
            gemm(abf, P(M8 + 1), P(M8 + 4), vtmp, G, EMB, EMB);       // W2 + b2
            bn(vtmp, P(M8 + 7), P(M8 + 5), G, EMB, true, vnb, nullptr, nullptr); // g2, b2n
        }
    }

    // ---- mean pooling + prediction head ----
    zero_f32<<<nblk((size_t)G, 256), 256, 0, stream>>>(cnt, G);
    count_batch<<<nblk((size_t)N, 256), 256, 0, stream>>>(node_batch, cnt, N);
    zero_f32<<<nblk((size_t)G * EMB, 256), 256, 0, stream>>>(s1, (size_t)G * EMB);
    seg_sum<<<nblk((size_t)N * 128, 256), 256, 0, stream>>>(h, node_batch, s1, N, EMB, 0);
    zero_f32<<<nblk((size_t)G * EMB, 256), 256, 0, stream>>>(s2, (size_t)G * EMB);
    seg_sum<<<nblk((size_t)Fn * 128, 256), 256, 0, stream>>>(face, fb, s2, Fn, EMB, 0);
    pool_build<<<nblk((size_t)G * EMB, 256), 256, 0, stream>>>(s1, s2, cnt, num_faces, aggG, G);
    f32_to_bf16_k<<<nblk((size_t)G * EMB2, 256), 256, 0, stream>>>(aggG, abf, (size_t)G * EMB2);
    gemm(abf, P(115), P(118), vtmp, G, EMB, EMB2);                    // pred W1 + b1
    bn(vtmp, P(120), P(117), G, EMB, true, vtmp, nullptr, nullptr);   // pred g, b
    final_matvec<<<nblk((size_t)G, 256), 256, 0, stream>>>(vtmp, P(116), P(119), (float*)d_out, G);
}
